// MockAttentionHead_77884936945679
// MI455X (gfx1250) — matrix operand order repaired and checked by reading, not compiled, not run on hardware
//
#include <hip/hip_runtime.h>
#include <hip/hip_bf16.h>

typedef __attribute__((ext_vector_type(16))) _Float16 v16h;
typedef __attribute__((ext_vector_type(8)))  float    v8f;
typedef __attribute__((ext_vector_type(4)))  int      v4i;
typedef __attribute__((ext_vector_type(4)))  float    v4f;

#define NPTS   8192
#define DIM    512
#define BM     128
#define BN     128
#define BK     32
#define LDH    40          // padded halves per LDS row (80 B, 16B-aligned)
#define NSTG   (DIM / BK)  // 16 K-stages
#define NCB    (NPTS / BN) // 64 column blocks
#define CMET   0.01f

#if defined(__has_builtin)
#if __has_builtin(__builtin_amdgcn_global_load_async_to_lds_b128)
#define USE_ASYNC_LDS 1
#endif
#if __has_builtin(__builtin_amdgcn_s_wait_asynccnt)
#define HAVE_WAIT_ASYNC_BUILTIN 1
#endif
#endif

#if defined(USE_ASYNC_LDS)
typedef __attribute__((address_space(1))) v4i g_v4i;   // global (AS1) int4*
typedef __attribute__((address_space(3))) v4i l_v4i;   // LDS (AS3) int4*
__device__ __forceinline__ void wait_async0() {
#if defined(HAVE_WAIT_ASYNC_BUILTIN)
    __builtin_amdgcn_s_wait_asynccnt(0);
#else
    asm volatile("s_wait_asynccnt 0x0" ::: "memory");
#endif
}
#endif

// ---------------- prep: f32 -> f16 copies + squared norms ----------------
__global__ void prep_kernel(const float* __restrict__ Q, const float* __restrict__ K,
                            _Float16* __restrict__ Q16, _Float16* __restrict__ K16,
                            float* __restrict__ qq, float* __restrict__ kk) {
    const int pt   = blockIdx.x;
    const int isK  = blockIdx.y;
    const int tid  = threadIdx.x;          // 256 threads
    const float* src = (isK ? K : Q) + (size_t)pt * DIM;
    _Float16*   dst = (isK ? K16 : Q16) + (size_t)pt * DIM;

    float v0 = src[tid];
    float v1 = src[tid + 256];
    dst[tid]       = (_Float16)v0;
    dst[tid + 256] = (_Float16)v1;

    float ss = v0 * v0 + v1 * v1;
    #pragma unroll
    for (int m = 16; m >= 1; m >>= 1) ss += __shfl_xor(ss, m, 32);

    __shared__ float ps[8];
    const int wave = tid >> 5, lane = tid & 31;
    if (lane == 0) ps[wave] = ss;
    __syncthreads();
    if (tid == 0) {
        float t = 0.f;
        #pragma unroll
        for (int i = 0; i < 8; ++i) t += ps[i];
        (isK ? kk : qq)[pt] = t;
    }
}

// ---------------- main fused GEMM + metric-distance + exp ----------------
__global__ void __launch_bounds__(256)
scores_kernel(const _Float16* __restrict__ Q16, const _Float16* __restrict__ K16,
              const float* __restrict__ qq, const float* __restrict__ kk,
              float* __restrict__ out, float* __restrict__ partial) {
#if defined(USE_ASYNC_LDS)
    __shared__ _Float16 lsA[2][BM * LDH];
    __shared__ _Float16 lsB[2][BN * LDH];
#else
    __shared__ _Float16 lsA[1][BM * LDH];
    __shared__ _Float16 lsB[1][BN * LDH];
#endif
    __shared__ float lsQQ[BM];
    __shared__ float lsKK[BN];
    __shared__ float lsRP[BM][2];

    const int tid  = threadIdx.x;
    const int wave = tid >> 5;
    const int lane = tid & 31;
    const int half = lane >> 4;
    const int l16  = lane & 15;
    const int waveM = wave & 3;   // 0..3 -> rows
    const int waveN = wave >> 2;  // 0..1 -> cols

    const int mBase = blockIdx.y * BM;
    const int nBase = blockIdx.x * BN;
    const int cb    = blockIdx.x;

    if (tid < BM) lsQQ[tid] = qq[mBase + tid];
    else if (tid < 2 * BM) lsKK[tid - BM] = kk[nBase + (tid - BM)];

    // global-load slots: 512 16B chunks per tile, 2 per thread
    const int c0 = tid, c1 = tid + 256;
    const int r0 = c0 >> 2, o0 = (c0 & 3) * 8;
    const int r1 = c1 >> 2, o1 = (c1 & 3) * 8;
    const _Float16* gA = Q16 + (size_t)(mBase) * DIM;
    const _Float16* gB = K16 + (size_t)(nBase) * DIM;

    v8f zero = {0.f, 0.f, 0.f, 0.f, 0.f, 0.f, 0.f, 0.f};
    v8f acc[2][4];
    #pragma unroll
    for (int t = 0; t < 2; ++t)
        #pragma unroll
        for (int u = 0; u < 4; ++u) acc[t][u] = zero;

    union Frag { uint4 u4[2]; v16h h; };

    const int aBase0 = (waveM * 32 + 0 * 16 + l16) * LDH + half * 8;
    const int aBase1 = (waveM * 32 + 1 * 16 + l16) * LDH + half * 8;

#if defined(USE_ASYNC_LDS)
    // -------- async (ASYNCcnt-tracked) global->LDS double-buffered pipeline --------
    auto issue_stage = [&](int ks, int buf) {
        const int k0 = ks * BK;
        __builtin_amdgcn_global_load_async_to_lds_b128(
            (g_v4i*)(gA + (size_t)r0 * DIM + k0 + o0), (l_v4i*)&lsA[buf][r0 * LDH + o0], 0, 0);
        __builtin_amdgcn_global_load_async_to_lds_b128(
            (g_v4i*)(gA + (size_t)r1 * DIM + k0 + o1), (l_v4i*)&lsA[buf][r1 * LDH + o1], 0, 0);
        __builtin_amdgcn_global_load_async_to_lds_b128(
            (g_v4i*)(gB + (size_t)r0 * DIM + k0 + o0), (l_v4i*)&lsB[buf][r0 * LDH + o0], 0, 0);
        __builtin_amdgcn_global_load_async_to_lds_b128(
            (g_v4i*)(gB + (size_t)r1 * DIM + k0 + o1), (l_v4i*)&lsB[buf][r1 * LDH + o1], 0, 0);
    };
    issue_stage(0, 0);
    wait_async0();
    __syncthreads();
    for (int ks = 0; ks < NSTG; ++ks) {
        const int buf = ks & 1;
        if (ks + 1 < NSTG) issue_stage(ks + 1, buf ^ 1);  // other buffer: no hazard with readers

        Frag fa[2];
        {
            const uint4* p0 = reinterpret_cast<const uint4*>(&lsA[buf][aBase0]);
            fa[0].u4[0] = p0[0]; fa[0].u4[1] = p0[2];
            const uint4* p1 = reinterpret_cast<const uint4*>(&lsA[buf][aBase1]);
            fa[1].u4[0] = p1[0]; fa[1].u4[1] = p1[2];
        }
        Frag fb[4];
        #pragma unroll
        for (int u = 0; u < 4; ++u) {
            const int bBase = (waveN * 64 + u * 16 + l16) * LDH + half * 8;
            const uint4* p = reinterpret_cast<const uint4*>(&lsB[buf][bBase]);
            fb[u].u4[0] = p[0]; fb[u].u4[1] = p[2];
        }
        #pragma unroll
        for (int t = 0; t < 2; ++t)
            #pragma unroll
            for (int u = 0; u < 4; ++u)
                acc[t][u] = __builtin_amdgcn_wmma_f32_16x16x32_f16(
                    false, fa[t].h, false, fb[u].h,
                    (short)0, acc[t][u], false, false);
        if (ks + 1 < NSTG) {
            wait_async0();     // next stage fully landed in LDS
            __syncthreads();   // all waves done reading current stage + see next stage
        }
    }
#else
    // -------- fallback: VGPR-staged global->LDS single-buffer pipeline --------
    for (int ks = 0; ks < NSTG; ++ks) {
        const int k0 = ks * BK;
        uint4 ra0 = *reinterpret_cast<const uint4*>(gA + (size_t)r0 * DIM + k0 + o0);
        uint4 ra1 = *reinterpret_cast<const uint4*>(gA + (size_t)r1 * DIM + k0 + o1);
        uint4 rb0 = *reinterpret_cast<const uint4*>(gB + (size_t)r0 * DIM + k0 + o0);
        uint4 rb1 = *reinterpret_cast<const uint4*>(gB + (size_t)r1 * DIM + k0 + o1);
        if (ks + 1 < NSTG) {
            __builtin_prefetch(gA + (size_t)r0 * DIM + k0 + BK + o0, 0, 3);
            __builtin_prefetch(gB + (size_t)r0 * DIM + k0 + BK + o0, 0, 3);
        }
        __syncthreads();
        *reinterpret_cast<uint4*>(&lsA[0][r0 * LDH + o0]) = ra0;
        *reinterpret_cast<uint4*>(&lsA[0][r1 * LDH + o1]) = ra1;
        *reinterpret_cast<uint4*>(&lsB[0][r0 * LDH + o0]) = rb0;
        *reinterpret_cast<uint4*>(&lsB[0][r1 * LDH + o1]) = rb1;
        __syncthreads();

        Frag fa[2];
        {
            const uint4* p0 = reinterpret_cast<const uint4*>(&lsA[0][aBase0]);
            fa[0].u4[0] = p0[0]; fa[0].u4[1] = p0[2];
            const uint4* p1 = reinterpret_cast<const uint4*>(&lsA[0][aBase1]);
            fa[1].u4[0] = p1[0]; fa[1].u4[1] = p1[2];
        }
        Frag fb[4];
        #pragma unroll
        for (int u = 0; u < 4; ++u) {
            const int bBase = (waveN * 64 + u * 16 + l16) * LDH + half * 8;
            const uint4* p = reinterpret_cast<const uint4*>(&lsB[0][bBase]);
            fb[u].u4[0] = p[0]; fb[u].u4[1] = p[2];
        }
        #pragma unroll
        for (int t = 0; t < 2; ++t)
            #pragma unroll
            for (int u = 0; u < 4; ++u)
                acc[t][u] = __builtin_amdgcn_wmma_f32_16x16x32_f16(
                    false, fa[t].h, false, fb[u].h,
                    (short)0, acc[t][u], false, false);
    }
#endif

    // ---------------- epilogue: metric distances, exp, row partials ----------------
    #pragma unroll
    for (int t = 0; t < 2; ++t) {
        #pragma unroll
        for (int r = 0; r < 8; ++r) {
            const int rowL = waveM * 32 + t * 16 + r + half * 8;
            const int row  = mBase + rowL;
            const float qv = lsQQ[rowL];
            float se = 0.f;
            #pragma unroll
            for (int u = 0; u < 4; ++u) {
                const int colL = waveN * 64 + u * 16 + l16;
                const float kv = lsKK[colL];
                const float s  = acc[t][u][r];
                float n2 = fmaxf(qv + kv - 2.f * s, 0.f);
                float qd = qv - s;
                float kd = s - kv;
                float dq = __builtin_sqrtf(n2 + CMET * qd * qd);
                float dk = __builtin_sqrtf(n2 + CMET * kd * kd);
                float e  = __expf(-0.5f * (dq + dk));   // score <= 0: never overflows
                // streaming output: non-temporal, don't thrash L2 (keep K16 resident)
                __builtin_nontemporal_store(e, &out[(size_t)row * NPTS + (nBase + colL)]);
                se += e;
            }
            // reduce across the 16 lanes of this half (masks stay inside half)
            #pragma unroll
            for (int m = 8; m >= 1; m >>= 1) se += __shfl_xor(se, m, 32);
            if (l16 == 0) lsRP[rowL][waveN] = se;
        }
    }
    __syncthreads();
    if (tid < BM) {
        // deterministic fixed-order combine of the two wave-columns
        partial[(size_t)cb * NPTS + mBase + tid] = lsRP[tid][0] + lsRP[tid][1];
    }
}

// ---------------- reduce partials -> per-row reciprocal sums ----------------
__global__ void rowreduce_kernel(const float* __restrict__ partial, float* __restrict__ rinv) {
    const int row = blockIdx.x * 256 + threadIdx.x;
    float s = 0.f;
    #pragma unroll 8
    for (int cb = 0; cb < NCB; ++cb) s += partial[(size_t)cb * NPTS + row];
    rinv[row] = 1.0f / s;
}

// ---------------- normalize in place (streaming, non-temporal) ----------------
__global__ void normalize_kernel(float* __restrict__ out, const float* __restrict__ rinv) {
    const size_t i = ((size_t)blockIdx.x * 256 + threadIdx.x) * 4;
    const int row = (int)(i >> 13);   // / 8192
    const float inv = rinv[row];
    v4f v = __builtin_nontemporal_load(reinterpret_cast<const v4f*>(out + i));
    v *= inv;
    __builtin_nontemporal_store(v, reinterpret_cast<v4f*>(out + i));
}

extern "C" void kernel_launch(void* const* d_in, const int* in_sizes, int n_in,
                              void* d_out, int out_size, void* d_ws, size_t ws_size,
                              hipStream_t stream) {
    (void)in_sizes; (void)n_in; (void)out_size; (void)ws_size;
    const float* Q = (const float*)d_in[0];
    const float* K = (const float*)d_in[1];
    float* out = (float*)d_out;

    char* ws = (char*)d_ws;
    _Float16* Q16 = (_Float16*)(ws);                         // 8 MiB
    _Float16* K16 = (_Float16*)(ws + (8u << 20));            // 8 MiB
    float* qq      = (float*)(ws + (16u << 20));             // 32 KiB
    float* kk      = (float*)(ws + (16u << 20) + 32768);     // 32 KiB
    float* partial = (float*)(ws + (16u << 20) + 65536);     // 2 MiB
    float* rinv    = (float*)(ws + (16u << 20) + 65536 + (2u << 20));

    prep_kernel<<<dim3(NPTS, 2), 256, 0, stream>>>(Q, K, Q16, K16, qq, kk);
    scores_kernel<<<dim3(NPTS / BN, NPTS / BM), 256, 0, stream>>>(Q16, K16, qq, kk, out, partial);
    rowreduce_kernel<<<NPTS / 256, 256, 0, stream>>>(partial, rinv);
    normalize_kernel<<<(NPTS * (NPTS / 4)) / 256, 256, 0, stream>>>(out, rinv);
}
